// AttentionBlock_35459249996222
// MI455X (gfx1250) — matrix operand order repaired and checked by reading, not compiled, not run on hardware
//
#include <hip/hip_runtime.h>
#include <stdint.h>

typedef __attribute__((ext_vector_type(16))) __bf16 bf16x16;
typedef __attribute__((ext_vector_type(8)))  __bf16 bf16x8;
typedef __attribute__((ext_vector_type(8)))  float  f32x8;

#define S_LEN    2048
#define DMODEL   2048
#define NHEADS   32
#define HDIM     64

union BF16Frag { bf16x16 v; bf16x8 h[2]; };

__device__ inline f32x8 zero8() { f32x8 z; for (int e = 0; e < 8; ++e) z[e] = 0.0f; return z; }

__device__ inline f32x8 wmma_bf16(bf16x16 a, bf16x16 b, f32x8 c) {
  return __builtin_amdgcn_wmma_f32_16x16x32_bf16(false, a, false, b, (short)0, c, false, false);
}

// ---- CDNA5 async global->LDS copy (ASYNCcnt-tracked), 16 bytes ----
typedef int v4i __attribute__((vector_size(16)));
typedef v4i __attribute__((address_space(1))) gas_v4i;   // global (AS1)
typedef v4i __attribute__((address_space(3))) las_v4i;   // LDS (AS3)

__device__ inline void async_copy_b128(const void* gsrc, void* ldsdst) {
#if __has_builtin(__builtin_amdgcn_global_load_async_to_lds_b128)
  __builtin_amdgcn_global_load_async_to_lds_b128(
      (gas_v4i*)(uintptr_t)gsrc, (las_v4i*)(uintptr_t)ldsdst, 0, 0);
#else
  const unsigned lds_off = (unsigned)(uintptr_t)ldsdst;   // low 32 bits = LDS byte offset
  asm volatile("global_load_async_to_lds_b128 %0, %1, off"
               :: "v"(lds_off), "v"(gsrc) : "memory");
#endif
}

__device__ inline void wait_async0() {
#if __has_builtin(__builtin_amdgcn_s_wait_asynccnt)
  __builtin_amdgcn_s_wait_asynccnt(0);
#else
  asm volatile("s_wait_asynccnt 0x0" ::: "memory");
#endif
}

// A-fragment 16x32 (ISA 7.12.2): lane<16 -> row=lane, K {0..7,16..23}; lane>=16 -> row=lane-16, K {8..15,24..31}
__device__ inline bf16x16 load_a_frag(const __bf16* base, int stride, int lane) {
  const int row  = lane & 15;
  const int koff = (lane < 16) ? 0 : 8;
  BF16Frag u;
  u.h[0] = *(const bf16x8*)(base + (size_t)row * stride + koff);
  u.h[1] = *(const bf16x8*)(base + (size_t)row * stride + koff + 16);
  return u.v;
}

// B-fragment 32x16: lane<16 -> col=lane, K {0..7,16..23}; lane>=16 -> col=lane-16, K {8..15,24..31}
__device__ inline bf16x16 load_b_frag(const __bf16* base, int stride, int lane) {
  const int col = lane & 15;
  const int kr  = (lane < 16) ? 0 : 8;
  BF16Frag u;
  u.h[0] = *(const bf16x8*)(base + (size_t)col * stride + kr);
  u.h[1] = *(const bf16x8*)(base + (size_t)col * stride + kr + 16);
  return u.v;
}

__global__ void cvt_f32_bf16(const float* __restrict__ in, __bf16* __restrict__ out, int n) {
  int i = blockIdx.x * blockDim.x + threadIdx.x;
  if (i < n) out[i] = (__bf16)in[i];
}

// C[M,N] = A[M,K] * B[N,K]^T   (both bf16 row-major; B rows = output channels)
// mode 0: write f32 C.  mode 1: Cout = residual + gate[n] * acc
// Double-buffered LDS fed by async global->LDS copies.
__global__ __launch_bounds__(256)
void gemm_bf16_wmma(const __bf16* __restrict__ A, const __bf16* __restrict__ B,
                    float* __restrict__ Cf32,
                    const float* __restrict__ residual, const float* __restrict__ gate,
                    float* __restrict__ Cout,
                    int M, int N, int K, int mode) {
  __shared__ __bf16 Asm[2][128 * 32];
  __shared__ __bf16 Bsm[2][128 * 32];
  const int tid  = threadIdx.x;
  const int lane = tid & 31;
  const int wave = tid >> 5;
  const int wm = wave >> 2;   // 0..1  -> 64-row group
  const int wn = wave & 3;    // 0..3  -> 32-col group
  const int m0 = blockIdx.y * 128;
  const int n0 = blockIdx.x * 128;

  const int ldrow = tid >> 1;
  const int ldcol = (tid & 1) * 16;

  f32x8 acc[4][2];
  for (int i = 0; i < 4; ++i) for (int j = 0; j < 2; ++j) acc[i][j] = zero8();

  // stage one 128x32 A-slab + 128x32 B-slab into buffer `buf`
  auto issue_slab = [&](int kb, int buf) {
    const __bf16* ag = A + (size_t)(m0 + ldrow) * K + kb + ldcol;
    const __bf16* bg = B + (size_t)(n0 + ldrow) * K + kb + ldcol;
    async_copy_b128(ag,     &Asm[buf][ldrow * 32 + ldcol]);
    async_copy_b128(ag + 8, &Asm[buf][ldrow * 32 + ldcol + 8]);
    async_copy_b128(bg,     &Bsm[buf][ldrow * 32 + ldcol]);
    async_copy_b128(bg + 8, &Bsm[buf][ldrow * 32 + ldcol + 8]);
  };

  const int nk = K >> 5;                 // K / 32
  issue_slab(0, 0);

  for (int it = 0; it < nk; ++it) {
    const int buf = it & 1;
    wait_async0();                       // slab `it` landed in LDS (this wave's asyncs)
    __syncthreads();                     // all waves' asyncs landed; prev reads of other buf done
    if (it + 1 < nk) issue_slab((it + 1) << 5, buf ^ 1);

    bf16x16 afrag[4];
    for (int i = 0; i < 4; ++i)
      afrag[i] = load_a_frag(&Asm[buf][(wm * 64 + i * 16) * 32], 32, lane);
    bf16x16 bfrag[2];
    for (int j = 0; j < 2; ++j)
      bfrag[j] = load_b_frag(&Bsm[buf][(wn * 32 + j * 16) * 32], 32, lane);

    for (int i = 0; i < 4; ++i)
      for (int j = 0; j < 2; ++j)
        acc[i][j] = wmma_bf16(afrag[i], bfrag[j], acc[i][j]);
    __syncthreads();                     // reads of `buf` complete before it is overwritten
  }

  // D layout: lane holds col n = lane&15; VGPR r -> row r (+8 for upper half)
  const int colL = lane & 15;
  const int rowH = (lane < 16) ? 0 : 8;
  for (int i = 0; i < 4; ++i) {
    for (int j = 0; j < 2; ++j) {
      const int n = n0 + wn * 32 + j * 16 + colL;
      for (int r = 0; r < 8; ++r) {
        const int m = m0 + wm * 64 + i * 16 + rowH + r;
        const size_t idx = (size_t)m * N + n;
        if (mode == 0) Cf32[idx] = acc[i][j][r];
        else           Cout[idx] = residual[idx] + gate[n] * acc[i][j][r];
      }
    }
  }
}

// qkv f32 [S][3*DMODEL] -> Qh/Kh bf16 [H][S][64] (RoPE applied), Vt bf16 [H][64][S]
__global__ void rope_pack(const float* __restrict__ qkv,
                          __bf16* __restrict__ Qh, __bf16* __restrict__ Kh,
                          __bf16* __restrict__ Vt) {
  const int s = blockIdx.x;
  const int h = blockIdx.y;
  const int d = threadIdx.x;                 // 0..63
  const size_t rowbase = (size_t)s * (3 * DMODEL) + h * HDIM;
  const float q = qkv[rowbase + d];
  const float k = qkv[rowbase + DMODEL + d];
  const float v = qkv[rowbase + 2 * DMODEL + d];
  const int dp = (d < 32) ? d + 32 : d - 32;
  const float qp = qkv[rowbase + dp];
  const float kp = qkv[rowbase + DMODEL + dp];
  const float sgn = (d < 32) ? -1.0f : 1.0f;
  const int j = d & 31;
  // inv_freq = 10000^(-2j/64) = exp(-(2j/64) * ln(10000))
  const float ang = (float)s * __expf(-(float)(2 * j) * (1.0f / 64.0f) * 9.210340371976184f);
  float sn, cs;
  __sincosf(ang, &sn, &cs);
  const size_t ho = (size_t)h * S_LEN * HDIM;
  Qh[ho + (size_t)s * HDIM + d] = (__bf16)(q * cs + sgn * qp * sn);
  Kh[ho + (size_t)s * HDIM + d] = (__bf16)(k * cs + sgn * kp * sn);
  Vt[((size_t)h * HDIM + d) * S_LEN + s] = (__bf16)v;
}

// Flash attention, causal. Block = 4 waves; each wave owns 16 q-rows; key tiles of 32.
__global__ __launch_bounds__(128)
void flash_attn_wmma(const __bf16* __restrict__ Qh, const __bf16* __restrict__ Kh,
                     const __bf16* __restrict__ Vt, __bf16* __restrict__ attnOut) {
  __shared__ __bf16 Plds[4][16 * 32];        // per-wave private P staging (D->A relayout)
  const int tid  = threadIdx.x;
  const int lane = tid & 31;
  const int wave = tid >> 5;
  const int h  = blockIdx.y;
  const int q0 = blockIdx.x * 64;
  const float scale = 0.125f;                // 1/sqrt(64)

  const __bf16* Qb = Qh + (size_t)h * S_LEN * HDIM;
  const __bf16* Kb = Kh + (size_t)h * S_LEN * HDIM;
  const __bf16* Vb = Vt + (size_t)h * HDIM * S_LEN;

  const int arow = q0 + wave * 16 + (lane & 15);
  const int koff = (lane < 16) ? 0 : 8;
  bf16x16 aq[2];
  for (int c = 0; c < 2; ++c) {
    BF16Frag u;
    u.h[0] = *(const bf16x8*)(Qb + (size_t)arow * HDIM + c * 32 + koff);
    u.h[1] = *(const bf16x8*)(Qb + (size_t)arow * HDIM + c * 32 + koff + 16);
    aq[c] = u.v;
  }

  f32x8 acc[4];
  for (int t = 0; t < 4; ++t) acc[t] = zero8();
  float mi[8], li[8];
  for (int r = 0; r < 8; ++r) { mi[r] = -1e30f; li[r] = 0.0f; }

  const int mrow0 = q0 + wave * 16 + ((lane < 16) ? 0 : 8); // VGPR r -> row mrow0+r
  const int keyn  = lane & 15;
  const int nkb   = (q0 + 64) / 32;

  for (int kb = 0; kb < nkb; ++kb) {
    const int k0 = kb * 32;
    // S = Q * K^T  (two 16-key column tiles, K-dim 64 in two chunks)
    f32x8 sfr[2];
    for (int t = 0; t < 2; ++t) {
      sfr[t] = zero8();
      const int keycol = k0 + t * 16 + (lane & 15);
      const int kr = (lane < 16) ? 0 : 8;
      for (int c = 0; c < 2; ++c) {
        BF16Frag u;
        u.h[0] = *(const bf16x8*)(Kb + (size_t)keycol * HDIM + c * 32 + kr);
        u.h[1] = *(const bf16x8*)(Kb + (size_t)keycol * HDIM + c * 32 + kr + 16);
        sfr[t] = wmma_bf16(aq[c], u.v, sfr[t]);
      }
    }
    // online softmax (row reductions within 16-lane halves)
    float alpha[8];
    for (int r = 0; r < 8; ++r) {
      const int m = mrow0 + r;
      float v0 = sfr[0][r] * scale; if (k0 + keyn      > m) v0 = -1e30f;
      float v1 = sfr[1][r] * scale; if (k0 + 16 + keyn > m) v1 = -1e30f;
      float mx = fmaxf(v0, v1);
      mx = fmaxf(mx, __shfl_xor(mx, 1, 32));
      mx = fmaxf(mx, __shfl_xor(mx, 2, 32));
      mx = fmaxf(mx, __shfl_xor(mx, 4, 32));
      mx = fmaxf(mx, __shfl_xor(mx, 8, 32));
      const float mnew = fmaxf(mi[r], mx);
      alpha[r] = __expf(mi[r] - mnew);
      const float p0 = __expf(v0 - mnew);
      const float p1 = __expf(v1 - mnew);
      const int mloc = ((lane < 16) ? 0 : 8) + r;
      Plds[wave][mloc * 32 + keyn]      = (__bf16)p0;
      Plds[wave][mloc * 32 + 16 + keyn] = (__bf16)p1;
      float ps = p0 + p1;
      ps += __shfl_xor(ps, 1, 32);
      ps += __shfl_xor(ps, 2, 32);
      ps += __shfl_xor(ps, 4, 32);
      ps += __shfl_xor(ps, 8, 32);
      li[r] = li[r] * alpha[r] + ps;
      mi[r] = mnew;
    }
    for (int t = 0; t < 4; ++t)
      for (int r = 0; r < 8; ++r)
        acc[t][r] *= alpha[r];
    // P back as A-fragment (DS ops in-order within wave; region is wave-private)
    bf16x16 ap;
    {
      BF16Frag u;
      const int m = lane & 15;
      u.h[0] = *(const bf16x8*)(&Plds[wave][m * 32 + koff]);
      u.h[1] = *(const bf16x8*)(&Plds[wave][m * 32 + koff + 16]);
      ap = u.v;
    }
    // O += P * V   (V transposed: keys contiguous per d-row)
    for (int t = 0; t < 4; ++t) {
      const int dcol = t * 16 + (lane & 15);
      const int kr = (lane < 16) ? 0 : 8;
      BF16Frag u;
      u.h[0] = *(const bf16x8*)(Vb + (size_t)dcol * S_LEN + k0 + kr);
      u.h[1] = *(const bf16x8*)(Vb + (size_t)dcol * S_LEN + k0 + kr + 16);
      acc[t] = wmma_bf16(ap, u.v, acc[t]);
    }
  }
  // epilogue: O / l_i  -> attn[s][h*64+d] bf16
  for (int t = 0; t < 4; ++t) {
    const int d = t * 16 + (lane & 15);
    for (int r = 0; r < 8; ++r) {
      const int m = mrow0 + r;
      attnOut[(size_t)m * DMODEL + h * HDIM + d] = (__bf16)(acc[t][r] / li[r]);
    }
  }
}

extern "C" void kernel_launch(void* const* d_in, const int* in_sizes, int n_in,
                              void* d_out, int out_size, void* d_ws, size_t ws_size,
                              hipStream_t stream) {
  const float* x    = (const float*)d_in[0];
  const float* Wqkv = (const float*)d_in[1];
  const float* Wo   = (const float*)d_in[2];
  const float* gate = (const float*)d_in[3];
  float* out = (float*)d_out;
  char* ws = (char*)d_ws;

  __bf16* xb    = (__bf16*)(ws + 0);                 //  8 MB: x bf16
  __bf16* wqkvb = (__bf16*)(ws + 8388608);           // 24 MB: Wqkv bf16
  __bf16* wob   = (__bf16*)(ws + 33554432);          //  8 MB: Wo bf16
  float*  qkv   = (float*) (ws + 41943040);          // 48 MB: qkv f32
  __bf16* QhP   = (__bf16*)(ws + 92274688);          //  8 MB: Q head-major (roped)
  __bf16* KhP   = (__bf16*)(ws + 100663296);         //  8 MB: K head-major (roped)
  __bf16* VtP   = (__bf16*)(ws + 109051904);         //  8 MB: V transposed
  __bf16* attnb = (__bf16*)(ws + 117440512);         //  8 MB: attention output bf16

  cvt_f32_bf16<<<(2048 * 2048 + 255) / 256, 256, 0, stream>>>(x, xb, 2048 * 2048);
  cvt_f32_bf16<<<(6144 * 2048 + 255) / 256, 256, 0, stream>>>(Wqkv, wqkvb, 6144 * 2048);
  cvt_f32_bf16<<<(2048 * 2048 + 255) / 256, 256, 0, stream>>>(Wo, wob, 2048 * 2048);

  // qkv[s][e] = sum_d x[s][d] * Wqkv[e][d]
  gemm_bf16_wmma<<<dim3(48, 16), 256, 0, stream>>>(xb, wqkvb, qkv,
                                                   nullptr, nullptr, nullptr,
                                                   2048, 6144, 2048, 0);
  rope_pack<<<dim3(2048, 32), 64, 0, stream>>>(qkv, QhP, KhP, VtP);
  flash_attn_wmma<<<dim3(32, 32), 128, 0, stream>>>(QhP, KhP, VtP, attnb);
  // out[s][e] = x[s][e] + gate[e] * sum_d attn[s][d] * Wo[e][d]
  gemm_bf16_wmma<<<dim3(16, 16), 256, 0, stream>>>(attnb, wob, nullptr,
                                                   x, gate, out,
                                                   2048, 2048, 2048, 1);
}
